// CFConv_83743272337867
// MI455X (gfx1250) — matrix operand order repaired and checked by reading, not compiled, not run on hardware
//
#include <hip/hip_runtime.h>
#include <hip/hip_bf16.h>

typedef __attribute__((ext_vector_type(16))) __bf16 v16bf;
typedef __attribute__((ext_vector_type(8)))  __bf16 v8bf;
typedef __attribute__((ext_vector_type(8)))  float  v8f;
typedef __attribute__((ext_vector_type(4)))  float  v4f;

#define RBF_DIM 128
#define DIM 64
#define WAVES_PER_BLOCK 8
#define BLOCK_THREADS 256
#define H_PAD 72   // LDS row stride in bf16 elements (144B) -> conflict-free b128 loads

__device__ __forceinline__ float softplus_b05(float x) {
  // (1/0.5)*log(1+exp(0.5*x)), linear when 0.5*x > 14
  float z  = 0.5f * x;
  float sp = 2.0f * __logf(1.0f + __expf(z));
  return (z > 14.0f) ? x : sp;
}

// Guaranteed native no-return f32 atomic add at device scope (L2 RMW).
__device__ __forceinline__ void atomic_add_f32(float* p, float v) {
  asm volatile("global_atomic_add_f32 %0, %1, off scope:SCOPE_DEV"
               :: "v"(p), "v"(v)
               : "memory");
}

__device__ __forceinline__ v16bf pack4(v4f a, v4f b, v4f c, v4f d) {
  v16bf r;
  #pragma unroll
  for (int i = 0; i < 4; ++i) {
    r[i]      = (__bf16)a[i];
    r[4 + i]  = (__bf16)b[i];
    r[8 + i]  = (__bf16)c[i];
    r[12 + i] = (__bf16)d[i];
  }
  return r;
}

// B-fragment (K x 16, bf16): element i<8 -> K = 8*(lane/16) + i,
// element i>=8 -> K = 16 + 8*(lane/16) + (i-8).
// p points at (W^T) column data for this lane, offset by 8*(lane/16).
__device__ __forceinline__ v16bf loadB16(const float* __restrict__ p) {
  const v4f* q = (const v4f*)p;
  return pack4(q[0], q[1], q[4], q[5]);
}

__global__ __launch_bounds__(BLOCK_THREADS) void cfconv_fused(
    const float* __restrict__ rbf, const float* __restrict__ nfeat,
    const int* __restrict__ src,  const int* __restrict__ dst,
    const float* __restrict__ W1, const float* __restrict__ b1,
    const float* __restrict__ W2, const float* __restrict__ b2,
    float* __restrict__ out, int nTiles, int totalWaves)
{
  __shared__ __align__(16) __bf16 hbuf[WAVES_PER_BLOCK][16][H_PAD];

  const int lane = threadIdx.x & 31;
  const int wv   = threadIdx.x >> 5;
  const int m    = lane & 15;   // row within 16 (edge for A, column for B/C)
  const int half = lane >> 4;   // lane-half selector in fragment layouts
  const int waveId = blockIdx.x * WAVES_PER_BLOCK + wv;

  for (int t = waveId; t < nTiles; t += totalWaves) {
    const int ebase = t << 4;

    // ---- GEMM1 A fragments: rbf tile [16 x 128] f32 -> 4x bf16 16x16x32 A-frags.
    // rbf is an 819MB single-use stream: non-temporal so it doesn't evict the
    // atomic destination / node_feats working set (25MB) from the 192MB L2.
    // 16-bit A layout: lanes 0-15: M=lane, elems 0..7 -> K=base+0..7,
    // elems 8..15 -> K=base+16..23; lanes 16-31 shift K by +8. base = kk*32.
    const float* rrow = rbf + (size_t)(ebase + m) * RBF_DIM;
    v16bf A1[4];
    #pragma unroll
    for (int kk = 0; kk < 4; ++kk) {
      const v4f* p0 = (const v4f*)(rrow + kk * 32 + 8 * half);
      const v4f* p1 = (const v4f*)(rrow + kk * 32 + 16 + 8 * half);
      v4f a = __builtin_nontemporal_load(p0);
      v4f b = __builtin_nontemporal_load(p0 + 1);
      v4f c = __builtin_nontemporal_load(p1);
      v4f d = __builtin_nontemporal_load(p1 + 1);
      A1[kk] = pack4(a, b, c, d);
    }

    // ---- Edge indices + row pointers (lane-half-uniform -> merged loads).
    const float* np[8];
    float*       op[8];
    #pragma unroll
    for (int i = 0; i < 8; ++i) {
      int e = ebase + i + 8 * half;
      np[i] = nfeat + (size_t)src[e] * DIM + m;
      op[i] = out   + (size_t)dst[e] * DIM + m;
    }

    // ---- GEMM1: h = softplus(rbf @ W1^T + b1); stash bf16 h-tile in LDS.
    // C layout: lane -> N (=output dim within ntile), VGPR i -> M = i + 8*half.
    #pragma unroll
    for (int nt = 0; nt < 4; ++nt) {
      v8f c1 = {0.f, 0.f, 0.f, 0.f, 0.f, 0.f, 0.f, 0.f};
      const int d = nt * 16 + m;                                // output dim
      const float* wrow = W1 + (size_t)d * RBF_DIM + half * 8;  // W1[d][r...]
      #pragma unroll
      for (int kk = 0; kk < 4; ++kk) {
        v16bf B = loadB16(wrow + kk * 32);
        c1 = __builtin_amdgcn_wmma_f32_16x16x32_bf16(
                 false, A1[kk], false, B, (short)0, c1, false, false);
      }
      const float bias = b1[d];
      #pragma unroll
      for (int i = 0; i < 8; ++i) {
        float y = softplus_b05(c1[i] + bias);
        hbuf[wv][i + 8 * half][d] = (__bf16)y;
      }
    }

    // LDS is in-order per wave; fence compiler + hardware before re-striping.
    asm volatile("s_wait_dscnt 0x0" ::: "memory");

    // ---- GEMM2 A fragments from LDS (re-stripe C layout -> A layout).
    const __bf16* hrow = &hbuf[wv][m][0];
    v16bf A2[2];
    #pragma unroll
    for (int kk = 0; kk < 2; ++kk) {
      v8bf lo = *(const v8bf*)(hrow + kk * 32 + 8 * half);
      v8bf hi = *(const v8bf*)(hrow + kk * 32 + 16 + 8 * half);
      v16bf a;
      #pragma unroll
      for (int i = 0; i < 8; ++i) { a[i] = lo[i]; a[i + 8] = hi[i]; }
      A2[kk] = a;
    }

    // ---- GEMM2 + bias + gather(node_feats[src]) * h2 + scatter-add to out[dst].
    #pragma unroll
    for (int nt = 0; nt < 4; ++nt) {
      v8f c2 = {0.f, 0.f, 0.f, 0.f, 0.f, 0.f, 0.f, 0.f};
      const int f = nt * 16 + m;                             // output dim
      const float* wrow2 = W2 + (size_t)f * DIM + half * 8;  // W2[f][d...]
      #pragma unroll
      for (int kk = 0; kk < 2; ++kk) {
        v16bf B = loadB16(wrow2 + kk * 32);
        c2 = __builtin_amdgcn_wmma_f32_16x16x32_bf16(
                 false, A2[kk], false, B, (short)0, c2, false, false);
      }
      const float bias = b2[f];
      #pragma unroll
      for (int i = 0; i < 8; ++i) {
        // nt*16 is a constant in the unrolled body: folds into the load/atomic
        // instruction offset instead of 64-bit VALU address math.
        float v = (c2[i] + bias) * np[i][nt * 16];
        atomic_add_f32(op[i] + nt * 16, v);
      }
    }
  }
}

__global__ void zero_f32(float* p, int n) {
  int i = blockIdx.x * blockDim.x + threadIdx.x;
  if (i < n) p[i] = 0.0f;
}

// Scalar fallback for E % 16 edges (not hit for E = 1.6M, kept for generality).
__global__ void cfconv_tail(const float* __restrict__ rbf, const float* __restrict__ nfeat,
                            const int* __restrict__ src, const int* __restrict__ dst,
                            const float* __restrict__ W1, const float* __restrict__ b1,
                            const float* __restrict__ W2, const float* __restrict__ b2,
                            float* __restrict__ out, int e0, int E) {
  int idx = blockIdx.x * blockDim.x + threadIdx.x;
  int total = (E - e0) * DIM;
  if (idx >= total) return;
  int e = e0 + idx / DIM;
  int f = idx % DIM;
  float acc = 0.f;
  for (int d = 0; d < DIM; ++d) {
    float s = 0.f;
    for (int r = 0; r < RBF_DIM; ++r)
      s += rbf[(size_t)e * RBF_DIM + r] * W1[(size_t)d * RBF_DIM + r];
    s = softplus_b05(s + b1[d]);
    acc += s * W2[(size_t)f * DIM + d];
  }
  float v = (acc + b2[f]) * nfeat[(size_t)src[e] * DIM + f];
  atomic_add_f32(out + (size_t)dst[e] * DIM + f, v);
}

extern "C" void kernel_launch(void* const* d_in, const int* in_sizes, int n_in,
                              void* d_out, int out_size, void* d_ws, size_t ws_size,
                              hipStream_t stream) {
  (void)n_in; (void)d_ws; (void)ws_size;
  const float* rbf   = (const float*)d_in[0];
  const float* nfeat = (const float*)d_in[1];
  const int*   src   = (const int*)d_in[2];
  const int*   dst   = (const int*)d_in[3];
  const float* W1    = (const float*)d_in[4];
  const float* b1    = (const float*)d_in[5];
  const float* W2    = (const float*)d_in[6];
  const float* b2    = (const float*)d_in[7];
  float* out = (float*)d_out;

  const int E      = in_sizes[2];
  const int nTiles = E >> 4;
  const int rem    = E & 15;

  // d_out is poisoned by the harness; the scatter is accumulate, so zero first.
  zero_f32<<<(out_size + 255) / 256, 256, 0, stream>>>(out, out_size);

  const int blocks     = 1024;                       // 8192 persistent waves
  const int totalWaves = blocks * WAVES_PER_BLOCK;
  if (nTiles > 0) {
    cfconv_fused<<<blocks, BLOCK_THREADS, 0, stream>>>(
        rbf, nfeat, src, dst, W1, b1, W2, b2, out, nTiles, totalWaves);
  }
  if (rem > 0) {
    int e0 = nTiles << 4;
    cfconv_tail<<<(rem * DIM + 255) / 256, 256, 0, stream>>>(
        rbf, nfeat, src, dst, W1, b1, W2, b2, out, e0, E);
  }
}
